// DeepseekMoE_60378650247253
// MI455X (gfx1250) — matrix operand order repaired and checked by reading, not compiled, not run on hardware
//
#include <hip/hip_runtime.h>
#include <hip/hip_bf16.h>
#include <math.h>

// ---------------- problem constants ----------------
#define TT 4096   // tokens = B*S
#define DD 512    // model dim
#define FF 2048   // ffn dim
#define EE 8      // routed experts
#define NSH 2     // shared experts

typedef __attribute__((ext_vector_type(16))) __bf16 bf16x16;
typedef __attribute__((ext_vector_type(8)))  __bf16 bf16x8;
typedef __attribute__((ext_vector_type(8)))  float  f32x8;
typedef __attribute__((ext_vector_type(4)))  unsigned u32x4;
typedef __attribute__((ext_vector_type(4)))  int   i32x4;
typedef __attribute__((ext_vector_type(8)))  int   i32x8;

// ---------------- helpers ----------------
__device__ __forceinline__ bf16x16 make_frag(bf16x8 lo, bf16x8 hi) {
    bf16x16 r;
#pragma unroll
    for (int i = 0; i < 8; ++i) { r[i] = lo[i]; r[i + 8] = hi[i]; }
    return r;
}

__device__ __forceinline__ float gelu_exact(float v) {
    return 0.5f * v * (1.0f + erff(v * 0.70710678118654752440f));
}

// Low 32 bits of a flat pointer into the LDS aperture == LDS byte offset.
__device__ __forceinline__ unsigned lds_byte_off(const void* p) {
    return (unsigned)(unsigned long long)(uintptr_t)p;
}

// Issue one TDM gather: 16 rows (16-bit indices) x 512 bf16 from `gbase`
// into LDS at `ldsoff`, padding +16B after every 1024B row (stride 520 elems).
__device__ __forceinline__ void tdm_gather16(unsigned long long gbase, unsigned ldsoff,
                                             const int* rows /*16 token ids*/) {
    u32x4 g0;
    g0[0] = 1u | (1u << 31);                         // count=1, gather_mode=1, 16-bit idx
    g0[1] = ldsoff;                                  // lds_addr (bytes)
    g0[2] = (unsigned)(gbase & 0xFFFFFFFFull);       // global_addr[31:0]
    g0[3] = (unsigned)(gbase >> 32) | (2u << 30);    // global_addr[56:32] | type=2

    i32x8 g1;
    g1[0] = (int)((1u << 16) |                       // data_size = 2 bytes
                  (1u << 20) |                       // pad_enable
                  (7u << 22) |                       // pad_interval: 256 DWORDs (=1024B)
                  (3u << 25));                       // pad_amount: 4 DWORDs (=16B)
    g1[1] = (int)(((unsigned)DD & 0xFFFFu) << 16);   // tensor_dim0[15:0] = 512
    g1[2] = (int)(((unsigned)TT & 0xFFFFu) << 16);   // tensor_dim0[31:16]=0 | tensor_dim1[15:0]=4096
    g1[3] = (int)(((unsigned)DD) << 16);             // tensor_dim1[31:16]=0 | tile_dim0 = 512
    g1[4] = 16;                                      // tile_dim1 = #valid indices
    g1[5] = DD;                                      // tensor_dim0_stride[31:0] = 512
    g1[6] = 0;
    g1[7] = 0;

    i32x4 g2, g3;
#pragma unroll
    for (int j = 0; j < 4; ++j) {
        unsigned lo = (unsigned)__builtin_amdgcn_readfirstlane(rows[2 * j]);
        unsigned hi = (unsigned)__builtin_amdgcn_readfirstlane(rows[2 * j + 1]);
        g2[j] = (int)(lo | (hi << 16));
    }
#pragma unroll
    for (int j = 0; j < 4; ++j) {
        unsigned lo = (unsigned)__builtin_amdgcn_readfirstlane(rows[8 + 2 * j]);
        unsigned hi = (unsigned)__builtin_amdgcn_readfirstlane(rows[8 + 2 * j + 1]);
        g3[j] = (int)(lo | (hi << 16));
    }

#if __clang_major__ >= 23
    i32x8 gz = {0, 0, 0, 0, 0, 0, 0, 0};
    __builtin_amdgcn_tensor_load_to_lds(g0, g1, g2, g3, gz, 0);
#else
    __builtin_amdgcn_tensor_load_to_lds(g0, g1, g2, g3, 0);
#endif
}

// ---------------- tiny utility kernels ----------------
__global__ void moe_zero_f32(float* p, int n) {
    int i = blockIdx.x * blockDim.x + threadIdx.x;
    int st = gridDim.x * blockDim.x;
    for (; i < n; i += st) p[i] = 0.0f;
}

__global__ void moe_zero_i32(int* p, int n) {
    int i = blockIdx.x * blockDim.x + threadIdx.x;
    if (i < n) p[i] = 0;
}

__global__ void moe_f32_to_bf16(const float* __restrict__ in, __bf16* __restrict__ out, int n) {
    int i = blockIdx.x * blockDim.x + threadIdx.x;
    int st = gridDim.x * blockDim.x;
    for (; i < n; i += st) out[i] = (__bf16)in[i];
}

// ---------------- gating: softmax + top-2 + bucket append ----------------
__global__ __launch_bounds__(256) void moe_gate(
    const float* __restrict__ x, const float* __restrict__ gw, const float* __restrict__ gb,
    int* __restrict__ cnt, int* __restrict__ tokidx, float* __restrict__ tokw) {
    __shared__ float gwl[EE][DD];                // 16 KB
    const int tid = threadIdx.x;
    for (int i = tid; i < EE * DD; i += 256) gwl[i / DD][i % DD] = gw[i];
    __syncthreads();

    const int t = blockIdx.x * 256 + tid;
    const float* xr = x + (size_t)t * DD;
    float l[EE];
#pragma unroll
    for (int e = 0; e < EE; ++e) l[e] = gb[e];
    for (int d = 0; d < DD; ++d) {
        const float xv = xr[d];
#pragma unroll
        for (int e = 0; e < EE; ++e) l[e] = fmaf(xv, gwl[e][d], l[e]);
    }
    float mx = l[0];
#pragma unroll
    for (int e = 1; e < EE; ++e) mx = fmaxf(mx, l[e]);
    float s = 0.0f;
#pragma unroll
    for (int e = 0; e < EE; ++e) { l[e] = expf(l[e] - mx); s += l[e]; }
    const float inv = 1.0f / s;
    int e0 = 0; float v0 = l[0];
#pragma unroll
    for (int e = 1; e < EE; ++e) if (l[e] > v0) { v0 = l[e]; e0 = e; }
    int e1 = -1; float v1 = -1.0f;
#pragma unroll
    for (int e = 0; e < EE; ++e) if (e != e0 && l[e] > v1) { v1 = l[e]; e1 = e; }

    int s0 = atomicAdd(&cnt[e0], 1);
    tokidx[e0 * TT + s0] = t; tokw[e0 * TT + s0] = v0 * inv;
    int s1 = atomicAdd(&cnt[e1], 1);
    tokidx[e1 * TT + s1] = t; tokw[e1 * TT + s1] = v1 * inv;
}

// ---------------- fused FFN: 32-token tiles, TDM-gathered activations ----------------
#define MT      32            // tokens per workgroup (2 WMMA m-tiles)
#define ASTRIDE (DD + 8)      // 520 -> 1040B row stride (TDM pad reproduces this)
#define FBLK    256           // F processed in 8 blocks; Hh stays small
#define HSTRIDE (FBLK + 8)    // 264

__global__ __launch_bounds__(256) void moe_ffn(
    const __bf16* __restrict__ xb,
    const __bf16* __restrict__ sw1b, const float* __restrict__ sb1,
    const __bf16* __restrict__ sw2b, const float* __restrict__ sb2,
    const __bf16* __restrict__ rw1b, const float* __restrict__ rb1,
    const __bf16* __restrict__ rw2b, const float* __restrict__ rb2,
    const int* __restrict__ cnt, const int* __restrict__ tokidx,
    const float* __restrict__ tokw, float* __restrict__ out) {

    __shared__ __bf16 Ah[MT][ASTRIDE];   // 32x512 bf16 activation tile (33,280 B)
    __shared__ __bf16 Hh[MT][HSTRIDE];   // 32x256 bf16 hidden tile    (16,896 B)
    __shared__ int   toks[MT];
    __shared__ float wts[MT];

    const int tid = threadIdx.x;
    const int bid = blockIdx.x;

    const __bf16* W1; const float* B1;
    const __bf16* W2; const float* B2;
    int tile; int e = 0; bool routed;

    if (bid < NSH * (TT / MT)) {
        const int inst = bid >> 7;          // TT/MT == 128
        tile = bid & 127;
        W1 = sw1b + (size_t)inst * FF * DD;  B1 = sb1 + inst * FF;
        W2 = sw2b + (size_t)inst * DD * FF;  B2 = sb2 + inst * DD;
        routed = false;
    } else {
        const int r = bid - NSH * (TT / MT);
        e = r >> 7;
        tile = r & 127;
        if (tile * MT >= cnt[e]) return;    // uniform early-exit before barriers
        W1 = rw1b + (size_t)e * FF * DD;     B1 = rb1 + e * FF;
        W2 = rw2b + (size_t)e * DD * FF;     B2 = rb2 + e * DD;
        routed = true;
    }

    if (tid < MT) {
        if (!routed) { toks[tid] = tile * MT + tid; wts[tid] = 1.0f; }
        else {
            const int j = tile * MT + tid;
            if (j < cnt[e]) { toks[tid] = tokidx[e * TT + j]; wts[tid] = tokw[e * TT + j]; }
            else            { toks[tid] = 0;                  wts[tid] = 0.0f; }
        }
    }
    __syncthreads();

    const int lane = tid & 31, wid = tid >> 5;
    const int ln   = lane & 15, half = lane >> 4;

    // ---- TDM gather: two 16-row descriptors stage the 32x512 tile into LDS ----
    if (wid == 0) {
        int rows[16];
        const unsigned long long gbase = (unsigned long long)(uintptr_t)xb;
#pragma unroll
        for (int i = 0; i < 16; ++i) rows[i] = toks[i];
        tdm_gather16(gbase, lds_byte_off(&Ah[0][0]), rows);
#pragma unroll
        for (int i = 0; i < 16; ++i) rows[i] = toks[16 + i];
        tdm_gather16(gbase, lds_byte_off(&Ah[16][0]), rows);
        __builtin_amdgcn_s_wait_tensorcnt(0);
    }
    __syncthreads();

    const f32x8 fzero = {0.f, 0.f, 0.f, 0.f, 0.f, 0.f, 0.f, 0.f};
    f32x8 acc2[4][2];
#pragma unroll
    for (int i = 0; i < 4; ++i) { acc2[i][0] = fzero; acc2[i][1] = fzero; }

    for (int fb = 0; fb < FF / FBLK; ++fb) {
        const int fbase = fb * FBLK;

        // ---- GEMM1: H[32,FBLK] = GELU(A * W1^T + b1); each B frag feeds 2 WMMAs ----
        for (int nt = wid; nt < FBLK / 16; nt += 8) {
            const __bf16* ap0 = &Ah[ln][half * 8];
            const __bf16* ap1 = &Ah[16 + ln][half * 8];
            const __bf16* bp  = W1 + (size_t)(fbase + nt * 16 + ln) * DD + half * 8;
            f32x8 acc0 = fzero, acc1 = fzero;
#pragma unroll
            for (int k0 = 0; k0 < DD; k0 += 32) {
                bf16x16 b  = make_frag(*(const bf16x8*)(bp + k0),
                                       *(const bf16x8*)(bp + k0 + 16));
                bf16x16 a0 = make_frag(*(const bf16x8*)(ap0 + k0),
                                       *(const bf16x8*)(ap0 + k0 + 16));
                bf16x16 a1 = make_frag(*(const bf16x8*)(ap1 + k0),
                                       *(const bf16x8*)(ap1 + k0 + 16));
                acc0 = __builtin_amdgcn_wmma_f32_16x16x32_bf16(
                    false, a0, false, b, (short)0, acc0, false, false);
                acc1 = __builtin_amdgcn_wmma_f32_16x16x32_bf16(
                    false, a1, false, b, (short)0, acc1, false, false);
            }
            const float bias = B1[fbase + nt * 16 + ln];
#pragma unroll
            for (int r = 0; r < 8; ++r) {
                const int m = r + half * 8;
                Hh[m][nt * 16 + ln]      = (__bf16)gelu_exact(acc0[r] + bias);
                Hh[16 + m][nt * 16 + ln] = (__bf16)gelu_exact(acc1[r] + bias);
            }
        }
        __syncthreads();

        // ---- GEMM2 partial: acc2 += H[32,FBLK] * W2^T; B frags shared by 2 m-tiles ----
#pragma unroll
        for (int i = 0; i < 4; ++i) {
            const int d0 = (wid + 8 * i) * 16;
            const __bf16* ap0 = &Hh[ln][half * 8];
            const __bf16* ap1 = &Hh[16 + ln][half * 8];
            const __bf16* bp  = W2 + (size_t)(d0 + ln) * FF + fbase + half * 8;
#pragma unroll
            for (int k0 = 0; k0 < FBLK; k0 += 32) {
                bf16x16 b  = make_frag(*(const bf16x8*)(bp + k0),
                                       *(const bf16x8*)(bp + k0 + 16));
                bf16x16 a0 = make_frag(*(const bf16x8*)(ap0 + k0),
                                       *(const bf16x8*)(ap0 + k0 + 16));
                bf16x16 a1 = make_frag(*(const bf16x8*)(ap1 + k0),
                                       *(const bf16x8*)(ap1 + k0 + 16));
                acc2[i][0] = __builtin_amdgcn_wmma_f32_16x16x32_bf16(
                    false, a0, false, b, (short)0, acc2[i][0], false, false);
                acc2[i][1] = __builtin_amdgcn_wmma_f32_16x16x32_bf16(
                    false, a1, false, b, (short)0, acc2[i][1], false, false);
            }
        }
        __syncthreads();   // before Hh is overwritten by the next F-block
    }

    // ---- epilogue: bias, per-row routing weight, atomic scatter-add ----
#pragma unroll
    for (int i = 0; i < 4; ++i) {
        const int dcol = (wid + 8 * i) * 16 + ln;
        const float bias = B2[dcol];
#pragma unroll
        for (int mt = 0; mt < 2; ++mt) {
#pragma unroll
            for (int r = 0; r < 8; ++r) {
                const int m = mt * 16 + r + half * 8;
                const float v = (acc2[i][mt][r] + bias) * wts[m];
                atomicAdd(&out[(size_t)toks[m] * DD + dcol], v);
            }
        }
    }
}

// ---------------- host-side launch ----------------
extern "C" void kernel_launch(void* const* d_in, const int* in_sizes, int n_in,
                              void* d_out, int out_size, void* d_ws, size_t ws_size,
                              hipStream_t stream) {
    (void)in_sizes; (void)n_in; (void)out_size; (void)ws_size;

    const float* x      = (const float*)d_in[0];
    const float* gate_w = (const float*)d_in[1];
    const float* gate_b = (const float*)d_in[2];
    const float* sw1    = (const float*)d_in[3];
    const float* sb1    = (const float*)d_in[4];
    const float* sw2    = (const float*)d_in[5];
    const float* sb2    = (const float*)d_in[6];
    const float* rw1    = (const float*)d_in[7];
    const float* rb1    = (const float*)d_in[8];
    const float* rw2    = (const float*)d_in[9];
    const float* rb2    = (const float*)d_in[10];
    float* out = (float*)d_out;

    char* ws = (char*)d_ws;
    size_t off = 0;
    auto carve = [&](size_t bytes) { void* p = ws + off; off = (off + bytes + 255) & ~(size_t)255; return p; };
    __bf16* xb   = (__bf16*)carve((size_t)TT * DD * 2);
    __bf16* sw1b = (__bf16*)carve((size_t)NSH * FF * DD * 2);
    __bf16* sw2b = (__bf16*)carve((size_t)NSH * DD * FF * 2);
    __bf16* rw1b = (__bf16*)carve((size_t)EE * FF * DD * 2);
    __bf16* rw2b = (__bf16*)carve((size_t)EE * DD * FF * 2);
    int*    cnt    = (int*)carve(EE * sizeof(int));
    int*    tokidx = (int*)carve((size_t)EE * TT * sizeof(int));
    float*  tokw   = (float*)carve((size_t)EE * TT * sizeof(float));

    moe_zero_f32<<<2048, 256, 0, stream>>>(out, TT * DD);
    moe_zero_i32<<<1, 32, 0, stream>>>(cnt, EE);

    moe_f32_to_bf16<<<1024, 256, 0, stream>>>(x,   xb,   TT * DD);
    moe_f32_to_bf16<<<1024, 256, 0, stream>>>(sw1, sw1b, NSH * FF * DD);
    moe_f32_to_bf16<<<1024, 256, 0, stream>>>(sw2, sw2b, NSH * DD * FF);
    moe_f32_to_bf16<<<2048, 256, 0, stream>>>(rw1, rw1b, EE * FF * DD);
    moe_f32_to_bf16<<<2048, 256, 0, stream>>>(rw2, rw2b, EE * DD * FF);

    moe_gate<<<TT / 256, 256, 0, stream>>>(x, gate_w, gate_b, cnt, tokidx, tokw);

    const int nblocks = (NSH + EE) * (TT / MT);   // 1280
    moe_ffn<<<nblocks, 256, 0, stream>>>(xb, sw1b, sb1, sw2b, sb2,
                                         rw1b, rb1, rw2b, rb2,
                                         cnt, tokidx, tokw, out);
}
